// SASRecStyleLayerBlock_27874337751169
// MI455X (gfx1250) — matrix operand order repaired and checked by reading, not compiled, not run on hardware
//
#include <hip/hip_runtime.h>

// ---------------------------------------------------------------------------
// CDNA5 (gfx1250) transformer encoder layer.
// bf16 end-to-end WMMA, TDM (tensor_load_to_lds) + async global->LDS staging.
// B=32, S=512 (T=16384 tokens), D=512, H=8, Dh=64, FFN=2048.
// ---------------------------------------------------------------------------

typedef __attribute__((ext_vector_type(16))) __bf16 v16bf;
typedef __attribute__((ext_vector_type(8)))  __bf16 v8bf;
typedef __attribute__((ext_vector_type(8)))  float  v8f;
typedef __attribute__((ext_vector_type(4)))  unsigned int v4u;
typedef __attribute__((ext_vector_type(8)))  int v8i;
typedef __attribute__((ext_vector_type(4)))  int v4i;

#define TOKENS 16384
#define DMODEL 512
#define NHEADS 8
#define HDIM   64
#define SEQ    512
#define DFF    2048

static __device__ __forceinline__ __bf16 f2bf(float f) { return (__bf16)f; }

static __device__ __forceinline__ v8f wmma_bf16(const v16bf& a, const v16bf& b, const v8f& c) {
    return __builtin_amdgcn_wmma_f32_16x16x32_bf16(false, a, false, b, (short)0, c, false, false);
}

// ---- async copy 16B global -> LDS (ASYNCcnt) ------------------------------
static __device__ __forceinline__ void async_ld_b128(unsigned int lds_off, const void* g) {
    asm volatile("global_load_async_to_lds_b128 %0, %1, off"
                 :: "v"(lds_off), "v"((unsigned long long)(size_t)g)
                 : "memory");
}
static __device__ __forceinline__ void wait_async0() {
    asm volatile("s_wait_asynccnt 0x0" ::: "memory");
}

// ---- TDM: 2D tile global -> LDS (TENSORcnt) -------------------------------
// D# per CDNA5 ISA. data_size=2B. Optional LDS row pad.
// 6-arg builtin form (clang-23 / therock headers): groups 0..3 + extra + cpol.
static __device__ __forceinline__ void tdm_load_2d(
    unsigned int lds_byte, const void* gaddr,
    unsigned int tensor_d0, unsigned int tensor_d1,      // tensor dims (elems)
    unsigned int tile_d0,   unsigned int tile_d1,        // tile dims (elems)
    unsigned int d0_stride_elems,
    unsigned int pad_interval_code, unsigned int pad_amount_code, bool pad_en)
{
    unsigned long long ga = (unsigned long long)(size_t)gaddr;
    v4u g0;
    g0[0] = 1u;                                            // count=1, user D#
    g0[1] = lds_byte;                                      // lds_addr
    g0[2] = (unsigned int)(ga & 0xFFFFFFFFull);            // global_addr[31:0]
    g0[3] = (unsigned int)((ga >> 32) & 0x1FFFFFFull)      // global_addr[56:32]
          | (2u << 30);                                    // type = 2 ("image")
    unsigned int w0 = (1u << 16);                          // data_size: 1 = 2 bytes
    if (pad_en) w0 |= (1u << 20) | (pad_interval_code << 22) | (pad_amount_code << 25);
    v8i g1;
    g1[0] = (int)w0;
    g1[1] = (int)((tensor_d0 & 0xFFFFu) << 16);            // dim0[15:0] (barrier addr = 0)
    g1[2] = (int)(((tensor_d0 >> 16) & 0xFFFFu) | ((tensor_d1 & 0xFFFFu) << 16));
    g1[3] = (int)(((tensor_d1 >> 16) & 0xFFFFu) | ((tile_d0 & 0xFFFFu) << 16));
    g1[4] = (int)(tile_d1 & 0xFFFFu);                      // tile_dim1 (tile_dim2 = 0)
    g1[5] = (int)d0_stride_elems;                          // dim0_stride[31:0]
    g1[6] = 0;                                             // dim0_stride[47:32], dim1_stride lo
    g1[7] = 0;
    v4i g2 = {0, 0, 0, 0};
    v4i g3 = {0, 0, 0, 0};
    v8i gx = {0, 0, 0, 0, 0, 0, 0, 0};
    __builtin_amdgcn_tensor_load_to_lds(g0, g1, g2, g3, gx, 0);
}

// concat two v8bf into a v16bf fragment
static __device__ __forceinline__ v16bf cat16(v8bf a, v8bf b) {
    return __builtin_shufflevector(a, b, 0,1,2,3,4,5,6,7,8,9,10,11,12,13,14,15);
}

// ---------------------------------------------------------------------------
// elementwise fp32 -> bf16 (weights / activations pre-pass)
// ---------------------------------------------------------------------------
__global__ __launch_bounds__(256) void cvt_bf16_k(const float* __restrict__ in,
                                                  __bf16* __restrict__ out) {
    int i = blockIdx.x * 256 + threadIdx.x;
    out[i] = f2bf(in[i]);
}

// ---------------------------------------------------------------------------
// bf16 WMMA GEMM: out = epilogue(A[M,K]bf16 @ W[K,N]bf16 + bias)
// MODE 1: scatter bf16 to [B,H,S,Dh]  (QKV)
// MODE 2: + residual fp32, fp32 out   (O-proj / FFN2)
// MODE 3: exact GELU, bf16 out        (FFN1)
// Block tile 128x64, BK=32, 8 waves, wave tile 32x32, double-buffered LDS.
// A tile staged by the Tensor Data Mover (one descriptor per tile, wave 0);
// W tile staged transposed ([n][k]) for contiguous B fragments.
// ---------------------------------------------------------------------------
#define LDA 40   // elems: 32 + 8 pad -> TDM pad: 64B data (code 3), 16B pad (code 3)
#define LDBT 40

template<int MODE>
__global__ __launch_bounds__(256) void gemm_bf16_wmma(
    const __bf16* __restrict__ A, const __bf16* __restrict__ W,
    const float* __restrict__ bias, const float* __restrict__ R,
    float* __restrict__ outF, __bf16* __restrict__ outB,
    int M, int N, int K)
{
    __shared__ __align__(16) __bf16 AsBuf[2][128 * LDA];
    __shared__ __align__(16) __bf16 BsBuf[2][64 * LDBT];

    const int tid   = threadIdx.x;
    const int wave  = tid >> 5;
    const int lane  = tid & 31;
    const int hf    = lane >> 4;
    const int ql    = lane & 15;
    const int waveM = wave >> 1;            // 0..3
    const int waveN = wave & 1;             // 0..1
    const int m0    = blockIdx.y * 128;
    const int n0    = blockIdx.x * 64;

    auto stageA = [&](int bb, int k0) {     // 128x32 bf16 tile via TDM (wave 0 only)
        tdm_load_2d((unsigned int)(size_t)&AsBuf[bb][0],
                    A + (size_t)m0 * K + k0,
                    /*tensor_d0=*/(unsigned)K, /*tensor_d1=*/(unsigned)M,
                    /*tile_d0=*/32, /*tile_d1=*/128,
                    /*d0_stride=*/(unsigned)K,
                    /*pad_interval=*/3, /*pad_amount=*/3, /*pad_en=*/true);
    };
    auto stageB = [&](int bb, int k0) {     // 32x64 tile, transposed to [n][k]
        int kk = tid >> 3, g = tid & 7;
        v8bf w = *(const v8bf*)(W + (size_t)(k0 + kk) * N + n0 + g * 8);
        __bf16* dst = &BsBuf[bb][0];
        #pragma unroll
        for (int j = 0; j < 8; ++j) dst[(g * 8 + j) * LDBT + kk] = w[j];
    };

    v8f acc[2][2];
    #pragma unroll
    for (int a = 0; a < 2; ++a)
        #pragma unroll
        for (int b = 0; b < 2; ++b) acc[a][b] = {};

    if (wave == 0) stageA(0, 0);
    stageB(0, 0);

    int buf = 0;
    for (int k0 = 0; k0 < K; k0 += 32) {
        const bool more = (k0 + 32 < K);
        if (more) {
            if (wave == 0) stageA(buf ^ 1, k0 + 32);
            stageB(buf ^ 1, k0 + 32);
        }
        if (wave == 0) {
            if (more) __builtin_amdgcn_s_wait_tensorcnt(1);
            else      __builtin_amdgcn_s_wait_tensorcnt(0);
        }
        __syncthreads();

        const __bf16* As = AsBuf[buf];
        const __bf16* Bs = BsBuf[buf];

        v16bf af[2], bfr[2];
        #pragma unroll
        for (int mt = 0; mt < 2; ++mt) {
            const __bf16* src = &As[(waveM * 32 + mt * 16 + ql) * LDA];
            af[mt] = cat16(*(const v8bf*)(src + hf * 8),
                           *(const v8bf*)(src + 16 + hf * 8));
        }
        #pragma unroll
        for (int nt = 0; nt < 2; ++nt) {
            const __bf16* src = &Bs[(waveN * 32 + nt * 16 + ql) * LDBT];
            bfr[nt] = cat16(*(const v8bf*)(src + hf * 8),
                            *(const v8bf*)(src + 16 + hf * 8));
        }
        #pragma unroll
        for (int mt = 0; mt < 2; ++mt)
            #pragma unroll
            for (int nt = 0; nt < 2; ++nt)
                acc[mt][nt] = wmma_bf16(af[mt], bfr[nt], acc[mt][nt]);
        __syncthreads();
        buf ^= 1;
    }

    #pragma unroll
    for (int mt = 0; mt < 2; ++mt)
        #pragma unroll
        for (int nt = 0; nt < 2; ++nt)
            #pragma unroll
            for (int j = 0; j < 8; ++j) {
                int m = m0 + waveM * 32 + mt * 16 + hf * 8 + j;
                int n = n0 + waveN * 32 + nt * 16 + ql;
                float v = acc[mt][nt][j] + bias[n];
                if (MODE == 2) {
                    v += R[(size_t)m * N + n];
                    outF[(size_t)m * N + n] = v;
                } else if (MODE == 3) {
                    v = 0.5f * v * (1.0f + erff(v * 0.70710678118654752f));
                    outB[(size_t)m * N + n] = f2bf(v);
                } else {  // MODE 1: QKV scatter
                    int b = m >> 9, s = m & (SEQ - 1);
                    int h = n >> 6, d = n & (HDIM - 1);
                    outB[(((size_t)(b * NHEADS + h) * SEQ + s) * HDIM) + d] = f2bf(v);
                }
            }
}

// ---------------------------------------------------------------------------
// Flash attention: Q,K,V bf16 [B,H,S,Dh]. Each wave owns 16 queries; the
// whole block shares one (b,h) key stream staged in LDS (K async, V^T manual).
// S^T = K Q^T  (softmax stats per-lane),  O^T = V^T P^T.
// ---------------------------------------------------------------------------
#define LDK 72   // elems: 64 + pad, row 144B
#define LDV 40   // elems: 32 + pad, row 80B

__global__ __launch_bounds__(256) void attention_wmma(
    const __bf16* __restrict__ Q, const __bf16* __restrict__ K,
    const __bf16* __restrict__ V, const float* __restrict__ mask,
    __bf16* __restrict__ ctx)
{
    __shared__ __align__(16) __bf16 Ks[32 * LDK];   // [key][d]
    __shared__ __align__(16) __bf16 Vs[64 * LDV];   // [d][key] (transposed)

    const int tid  = threadIdx.x;
    const int wave = tid >> 5;
    const int lane = tid & 31;
    const int hf   = lane >> 4;
    const int ql   = lane & 15;

    const int bh = blockIdx.x >> 2;                 // b*8+h
    const int qc = blockIdx.x & 3;
    const int b  = bh >> 3;
    const int q  = qc * 128 + wave * 16 + ql;

    const __bf16* Qp = Q + (size_t)bh * SEQ * HDIM;
    const __bf16* Kp = K + (size_t)bh * SEQ * HDIM;
    const __bf16* Vp = V + (size_t)bh * SEQ * HDIM;
    const float*  Mp = mask + ((size_t)b * SEQ + q) * SEQ;

    // Q^T B-fragments (lane = q): elem i -> d = s*32 + 16*hf + i
    v16bf bq[2];
    #pragma unroll
    for (int s = 0; s < 2; ++s) {
        const __bf16* qp = Qp + (size_t)q * HDIM + s * 32 + hf * 16;
        bq[s] = cat16(*(const v8bf*)qp, *(const v8bf*)(qp + 8));
    }

    v8f accO[4];
    #pragma unroll
    for (int dt = 0; dt < 4; ++dt) accO[dt] = {};
    float rmax = -3.0e38f, rsum = 0.0f;

    const unsigned int kbase = (unsigned int)(size_t)&Ks[0];

    for (int kb = 0; kb < SEQ; kb += 32) {
        // ---- stage K block (32x64) via async DMA ----
        {
            int r = tid >> 3, sg = tid & 7;
            async_ld_b128(kbase + r * (LDK * 2) + sg * 16,
                          Kp + (size_t)(kb + r) * HDIM + sg * 8);
        }
        // ---- stage V block transposed ([d][key]) ----
        {
            int ky = tid >> 3, g = tid & 7;
            v8bf vv = *(const v8bf*)(Vp + (size_t)(kb + ky) * HDIM + g * 8);
            #pragma unroll
            for (int j = 0; j < 8; ++j) Vs[(g * 8 + j) * LDV + ky] = vv[j];
        }
        wait_async0();
        __syncthreads();

        // ---- S^T = K(32xDh) @ Q^T(Dhx16): two 16-key M-tiles ----
        v8f sc[2];
        #pragma unroll
        for (int t = 0; t < 2; ++t) {
            sc[t] = {};
            #pragma unroll
            for (int s = 0; s < 2; ++s) {
                const __bf16* kr = &Ks[(t * 16 + ql) * LDK + s * 32 + hf * 16];
                v16bf ak = cat16(*(const v8bf*)kr, *(const v8bf*)(kr + 8));
                sc[t] = wmma_bf16(ak, bq[s], sc[t]);
            }
        }

        // ---- scale + mask, online softmax (lane = query) ----
        float p[2][8], bm = -3.0e38f;
        #pragma unroll
        for (int t = 0; t < 2; ++t)
            #pragma unroll
            for (int j = 0; j < 8; ++j) {
                float v = sc[t][j] * 0.125f + Mp[kb + t * 16 + hf * 8 + j];
                p[t][j] = v;
                bm = fmaxf(bm, v);
            }
        bm = fmaxf(bm, __shfl_xor(bm, 16, 32));
        float nmax = fmaxf(rmax, bm);
        float corr = __expf(rmax - nmax);
        rmax = nmax;

        float psum = 0.0f;
        #pragma unroll
        for (int t = 0; t < 2; ++t)
            #pragma unroll
            for (int j = 0; j < 8; ++j) {
                p[t][j] = __expf(p[t][j] - rmax);
                psum += p[t][j];
            }
        psum += __shfl_xor(psum, 16, 32);
        rsum = rsum * corr + psum;
        #pragma unroll
        for (int dt = 0; dt < 4; ++dt)
            #pragma unroll
            for (int j = 0; j < 8; ++j) accO[dt][j] *= corr;

        // ---- P^T B-fragment (elem i -> key = i + 16*hf): one half-exchange ----
        float ex0[8], ex1[8];
        #pragma unroll
        for (int j = 0; j < 8; ++j) {
            ex0[j] = __shfl_xor(p[0][j], 16, 32);
            ex1[j] = __shfl_xor(p[1][j], 16, 32);
        }
        v16bf pb;
        #pragma unroll
        for (int i = 0; i < 8; ++i) {
            pb[i]     = f2bf(hf ? ex1[i] : p[0][i]);
            pb[i + 8] = f2bf(hf ? p[1][i] : ex0[i]);
        }

        // ---- O^T += V^T @ P^T ----
        #pragma unroll
        for (int dt = 0; dt < 4; ++dt) {
            const __bf16* vr = &Vs[(dt * 16 + ql) * LDV + hf * 16];
            v16bf av = cat16(*(const v8bf*)vr, *(const v8bf*)(vr + 8));
            accO[dt] = wmma_bf16(av, pb, accO[dt]);
        }
        __syncthreads();
    }

    // ---- ctx[b, q, h*64+d] bf16 (token-major for O-projection) ----
    const int h = bh & 7;
    const float inv = 1.0f / rsum;
    #pragma unroll
    for (int dt = 0; dt < 4; ++dt)
        #pragma unroll
        for (int j = 0; j < 8; ++j) {
            int d = dt * 16 + hf * 8 + j;
            ctx[((size_t)b * SEQ + q) * DMODEL + h * HDIM + d] = f2bf(accO[dt][j] * inv);
        }
}

// ---------------------------------------------------------------------------
// Row LayerNorm over D=512, one wave per row. Optionally also emits a bf16
// copy (feeds the next GEMM's A operand).
// ---------------------------------------------------------------------------
template<bool WB>
__global__ __launch_bounds__(256) void layernorm_k(
    const float* __restrict__ x, const float* __restrict__ g,
    const float* __restrict__ bta, float* __restrict__ outF,
    __bf16* __restrict__ outB)
{
    const int wave = threadIdx.x >> 5;
    const int lane = threadIdx.x & 31;
    const int row  = blockIdx.x * 8 + wave;
    const float* xr = x + (size_t)row * DMODEL;

    float v[16], s = 0.0f;
    #pragma unroll
    for (int i = 0; i < 16; ++i) { v[i] = xr[lane + i * 32]; s += v[i]; }
    #pragma unroll
    for (int off = 16; off >= 1; off >>= 1) s += __shfl_xor(s, off, 32);
    const float mu = s * (1.0f / DMODEL);

    float var = 0.0f;
    #pragma unroll
    for (int i = 0; i < 16; ++i) { float d = v[i] - mu; var += d * d; }
    #pragma unroll
    for (int off = 16; off >= 1; off >>= 1) var += __shfl_xor(var, off, 32);
    const float r = rsqrtf(var * (1.0f / DMODEL) + 1e-12f);

    #pragma unroll
    for (int i = 0; i < 16; ++i) {
        int c = lane + i * 32;
        float o = (v[i] - mu) * r * g[c] + bta[c];
        outF[(size_t)row * DMODEL + c] = o;
        if (WB) outB[(size_t)row * DMODEL + c] = f2bf(o);
    }
}

// ---------------------------------------------------------------------------
extern "C" void kernel_launch(void* const* d_in, const int* in_sizes, int n_in,
                              void* d_out, int out_size, void* d_ws, size_t ws_size,
                              hipStream_t stream)
{
    const float* x    = (const float*)d_in[0];
    const float* msk  = (const float*)d_in[1];
    const float* Wq   = (const float*)d_in[2];
    const float* bq   = (const float*)d_in[3];
    const float* Wk   = (const float*)d_in[4];
    const float* bk   = (const float*)d_in[5];
    const float* Wv   = (const float*)d_in[6];
    const float* bv   = (const float*)d_in[7];
    const float* Wo   = (const float*)d_in[8];
    const float* bo   = (const float*)d_in[9];
    const float* ln1g = (const float*)d_in[10];
    const float* ln1b = (const float*)d_in[11];
    const float* W1   = (const float*)d_in[12];
    const float* b1   = (const float*)d_in[13];
    const float* W2   = (const float*)d_in[14];
    const float* b2   = (const float*)d_in[15];
    const float* ln2g = (const float*)d_in[16];
    const float* ln2b = (const float*)d_in[17];

    const size_t TD = (size_t)TOKENS * DMODEL;      // 8.39M
    const size_t TF = (size_t)TOKENS * DFF;         // 33.5M
    const size_t WD = (size_t)DMODEL * DMODEL;      // 262K
    const size_t WF = (size_t)DMODEL * DFF;         // 1.05M

    // bf16 region
    __bf16* bp  = (__bf16*)d_ws;
    __bf16* Xb  = bp;             bp += TD;
    __bf16* Qb  = bp;             bp += TD;
    __bf16* Kb  = bp;             bp += TD;
    __bf16* Vb  = bp;             bp += TD;
    __bf16* Cxb = bp;             bp += TD;
    __bf16* AOb = bp;             bp += TD;
    __bf16* Hb  = bp;             bp += TF;
    __bf16* WqB = bp;             bp += WD;
    __bf16* WkB = bp;             bp += WD;
    __bf16* WvB = bp;             bp += WD;
    __bf16* WoB = bp;             bp += WD;
    __bf16* W1B = bp;             bp += WF;
    __bf16* W2B = bp;             bp += WF;
    // fp32 region (bf16 counts above are even -> 4B aligned)
    float* fp  = (float*)bp;
    float* Y   = fp;              fp += TD;
    float* AOf = fp;              fp += TD;

    dim3 blk(256);
    dim3 gD(DMODEL / 64, TOKENS / 128);
    dim3 gF(DFF / 64,    TOKENS / 128);

    // 0) pre-convert weights + input activations to bf16
    cvt_bf16_k<<<dim3(TD / 256), blk, 0, stream>>>(x,  Xb);
    cvt_bf16_k<<<dim3(WD / 256), blk, 0, stream>>>(Wq, WqB);
    cvt_bf16_k<<<dim3(WD / 256), blk, 0, stream>>>(Wk, WkB);
    cvt_bf16_k<<<dim3(WD / 256), blk, 0, stream>>>(Wv, WvB);
    cvt_bf16_k<<<dim3(WD / 256), blk, 0, stream>>>(Wo, WoB);
    cvt_bf16_k<<<dim3(WF / 256), blk, 0, stream>>>(W1, W1B);
    cvt_bf16_k<<<dim3(WF / 256), blk, 0, stream>>>(W2, W2B);

    // 1) QKV projections -> bf16 [B,H,S,Dh]
    gemm_bf16_wmma<1><<<gD, blk, 0, stream>>>(Xb, WqB, bq, nullptr, nullptr, Qb, TOKENS, DMODEL, DMODEL);
    gemm_bf16_wmma<1><<<gD, blk, 0, stream>>>(Xb, WkB, bk, nullptr, nullptr, Kb, TOKENS, DMODEL, DMODEL);
    gemm_bf16_wmma<1><<<gD, blk, 0, stream>>>(Xb, WvB, bv, nullptr, nullptr, Vb, TOKENS, DMODEL, DMODEL);

    // 2) attention -> ctx bf16 [B,S,D]
    attention_wmma<<<dim3(32 * NHEADS * (SEQ / 128)), blk, 0, stream>>>(Qb, Kb, Vb, msk, Cxb);

    // 3) O-projection + residual(x) -> Y fp32 ; LN1 -> AOf + AOb
    gemm_bf16_wmma<2><<<gD, blk, 0, stream>>>(Cxb, WoB, bo, x, Y, nullptr, TOKENS, DMODEL, DMODEL);
    layernorm_k<true><<<dim3(TOKENS / 8), blk, 0, stream>>>(Y, ln1g, ln1b, AOf, AOb);

    // 4) FFN1 + GELU -> Hb bf16
    gemm_bf16_wmma<3><<<gF, blk, 0, stream>>>(AOb, W1B, b1, nullptr, nullptr, Hb, TOKENS, DFF, DMODEL);

    // 5) FFN2 + residual(AOf) -> Y fp32 ; LN2 -> d_out fp32
    gemm_bf16_wmma<2><<<gD, blk, 0, stream>>>(Hb, W2B, b2, AOf, Y, nullptr, TOKENS, DMODEL, DFF);
    layernorm_k<false><<<dim3(TOKENS / 8), blk, 0, stream>>>(Y, ln2g, ln2b, (float*)d_out, nullptr);
}
